// BasicModel_47107201303294
// MI455X (gfx1250) — matrix operand order repaired and checked by reading, not compiled, not run on hardware
//
#include <hip/hip_runtime.h>
#include <hip/hip_bf16.h>
#include <stdint.h>

typedef __attribute__((ext_vector_type(16))) __bf16 v16bf;
typedef __attribute__((ext_vector_type(8)))  float  v8f;

#define N_NODES  50000
#define F_IN     128
#define HIDDEN   256
#define N_GRAPHS 64

#define KCHUNK 128
#define APAD   8
#define AROW   (KCHUNK + APAD)    // 136 bf16 elements per LDS row (272B, conflict-free)

// ---------- helpers ----------
__device__ __forceinline__ uint16_t f2bf(float f) {
  uint32_t u = __float_as_uint(f);
  uint32_t r = u + 0x7FFFu + ((u >> 16) & 1u);   // round-to-nearest-even
  return (uint16_t)(r >> 16);
}
__device__ __forceinline__ float bf2f(uint16_t h) {
  return __uint_as_float(((uint32_t)h) << 16);
}

// ---------- elementwise conversions ----------
__global__ void k_f2bf(const float* __restrict__ in, uint16_t* __restrict__ out, long n) {
  long i = (long)blockIdx.x * blockDim.x + threadIdx.x;
  if (i < n) out[i] = f2bf(in[i]);
}

// W [K,N] f32 row-major -> Wt [N,K] bf16 row-major (B-fragment loads become contiguous)
__global__ void k_transpose_bf(const float* __restrict__ W, uint16_t* __restrict__ Wt,
                               int K, int N) {
  long i = (long)blockIdx.x * blockDim.x + threadIdx.x;
  if (i >= (long)K * N) return;
  int k = (int)(i / N), n = (int)(i % N);
  Wt[(long)n * K + k] = f2bf(W[i]);
}

__global__ void k_fill_u32(uint32_t* __restrict__ p, uint32_t val, long n) {
  long i = (long)blockIdx.x * blockDim.x + threadIdx.x;
  if (i < n) p[i] = val;
}

// ---------- WMMA GEMM ----------
// O[M,N](bf16) = act(A[M,K](bf16) @ W[K,N] + bias), Wt = W^T in [N,K] bf16.
// Block = 128 threads (4 waves) -> one 16x256 output slab.
// A slab (16 x KCHUNK) double-buffered in LDS via async global->LDS copies
// (ASYNCcnt), shared by all 4 waves. Each wave computes 16x64 = 4 WMMA tiles.
// Per K-step: gather A frag + all 4 B frags, pin them live via a scalar-operand
// asm barrier, then 4 back-to-back WMMAs. Requires N%256==0, K%128==0.
union FragBF { uint4 u[2]; v16bf v; };

__global__ void __launch_bounds__(128)
k_gemm_wmma(const uint16_t* __restrict__ A, int lda,
            const uint16_t* __restrict__ Wt,
            const float* __restrict__ bias,
            uint16_t* __restrict__ O, int ldo,
            int M, int N, int K, int act) {
  __shared__ uint16_t sA[2 * 16 * AROW];

  const int tid  = threadIdx.x;
  const int lane = tid & 31;
  const int wave = tid >> 5;                 // 0..3
  const int nslabs = N >> 8;                 // N / 256
  const int bm = blockIdx.x / nslabs;
  const int bn = (blockIdx.x % nslabs) << 8;
  if (bm >= (M >> 4)) return;                // uniform across block
  const long rowBase = (long)bm * 16;

  // async copy of one 16 x KCHUNK slab of A into LDS buffer `buf`
  auto issue_chunk = [&](int buf, int kbase) {
#pragma unroll
    for (int c = tid; c < 256; c += 128) {   // 256 x 16B chunks, 2 per thread
      int row = c >> 4, seg = c & 15;
      uint32_t lds_off = (uint32_t)(uintptr_t)&sA[buf * 16 * AROW + row * AROW + seg * 8];
      const uint16_t* g = A + (rowBase + row) * lda + kbase + seg * 8;
      asm volatile("global_load_async_to_lds_b128 %0, %1, off"
                   :: "v"(lds_off), "v"(g) : "memory");
    }
  };

  // B fragment pointers for this wave's 4 N-tiles (lane n = lane%16;
  // lanes 0-15 hold K 0..15, lanes 16-31 hold K 16..31 of each K-step)
  const int ncol = bn + wave * 64;
  const long kgB = (lane < 16) ? 0 : 16;
  const uint16_t* b0 = Wt + (long)(ncol +  0 + (lane & 15)) * K + kgB;
  const uint16_t* b1 = Wt + (long)(ncol + 16 + (lane & 15)) * K + kgB;
  const uint16_t* b2 = Wt + (long)(ncol + 32 + (lane & 15)) * K + kgB;
  const uint16_t* b3 = Wt + (long)(ncol + 48 + (lane & 15)) * K + kgB;

  v8f acc0 = {}, acc1 = {}, acc2 = {}, acc3 = {};

  issue_chunk(0, 0);
  int cur = 0;
  for (int kc = 0; kc < K; kc += KCHUNK) {
    asm volatile("s_wait_asynccnt 0x0" ::: "memory");   // own async copies done
    __syncthreads();                                    // whole slab visible
    if (kc + KCHUNK < K) issue_chunk(cur ^ 1, kc + KCHUNK);

    // A fragment rows from LDS: lane m = lane%16; lanes 0-15 hold K 0..15,
    // lanes 16-31 hold K 16..31 of each 32-wide K-step
    const uint16_t* aL = &sA[cur * 16 * AROW + (lane & 15) * AROW + ((lane < 16) ? 0 : 16)];
#pragma unroll
    for (int kk = 0; kk < KCHUNK; kk += 32) {
      // --- gather phase: A frag (LDS) + all four B frags (global, L2-hot) ---
      FragBF a, f0, f1, f2, f3;
      a.u[0]  = *(const uint4*)(aL + kk);
      a.u[1]  = *(const uint4*)(aL + kk + 8);
      f0.u[0] = *(const uint4*)(b0 + kc + kk);
      f0.u[1] = *(const uint4*)(b0 + kc + kk + 8);
      f1.u[0] = *(const uint4*)(b1 + kc + kk);
      f1.u[1] = *(const uint4*)(b1 + kc + kk + 8);
      f2.u[0] = *(const uint4*)(b2 + kc + kk);
      f2.u[1] = *(const uint4*)(b2 + kc + kk + 8);
      f3.u[0] = *(const uint4*)(b3 + kc + kk);
      f3.u[1] = *(const uint4*)(b3 + kc + kk + 8);
      // liveness barrier using scalar (32-bit) operands only — this backend
      // rejects vector-typed asm operands. Consuming one lane of each load
      // result forces all 10 loads to retire (into distinct VGPR tuples)
      // before any WMMA issues, batching loads ahead of the compute phase.
      asm volatile(""
                   :: "v"(a.u[0].x),  "v"(a.u[1].x),
                      "v"(f0.u[0].x), "v"(f0.u[1].x),
                      "v"(f1.u[0].x), "v"(f1.u[1].x),
                      "v"(f2.u[0].x), "v"(f2.u[1].x),
                      "v"(f3.u[0].x), "v"(f3.u[1].x));
      // --- compute phase: 4 pipelined WMMAs sharing the A fragment ---
      acc0 = __builtin_amdgcn_wmma_f32_16x16x32_bf16(false, a.v, false, f0.v, (short)0, acc0, false, false);
      acc1 = __builtin_amdgcn_wmma_f32_16x16x32_bf16(false, a.v, false, f1.v, (short)0, acc1, false, false);
      acc2 = __builtin_amdgcn_wmma_f32_16x16x32_bf16(false, a.v, false, f2.v, (short)0, acc2, false, false);
      acc3 = __builtin_amdgcn_wmma_f32_16x16x32_bf16(false, a.v, false, f3.v, (short)0, acc3, false, false);
    }
    __syncthreads();                                    // done reading `cur`
    cur ^= 1;
  }

  // C/D layout: VGPR i -> row (lane<16 ? i : i+8), col = lane%16
  const int rbase = (int)rowBase + ((lane >> 4) << 3);
  const int cb = ncol + (lane & 15);
  auto store_tile = [&](v8f acc, int colg) {
    float bv = bias ? bias[colg] : 0.0f;
#pragma unroll
    for (int i = 0; i < 8; ++i) {
      float v = acc[i] + bv;
      if (act) v = tanhf(v);
      O[(long)(rbase + i) * ldo + colg] = f2bf(v);
    }
  };
  store_tile(acc0, cb);
  store_tile(acc1, cb + 16);
  store_tile(acc2, cb + 32);
  store_tile(acc3, cb + 48);
}

// ---------- SpMM: agg[dst] += w * t[src], one wave per edge, 8 feats/lane ----------
__global__ void k_spmm(const int* __restrict__ src, const int* __restrict__ dst,
                       const float* __restrict__ w, const uint16_t* __restrict__ t,
                       float* __restrict__ agg, int E) {
  int lane = threadIdx.x & 31;
  int wave = threadIdx.x >> 5;
  int e = blockIdx.x * (blockDim.x >> 5) + wave;
  if (e >= E) return;
  int s = src[e], d = dst[e];
  float ew = w[e];
  const uint16_t* trow = t + (long)s * HIDDEN + lane * 8;
  float* arow = agg + (long)d * HIDDEN + lane * 8;
  uint4 pk = *(const uint4*)trow;                      // 8 bf16, one b128 load
  const uint16_t* ph = (const uint16_t*)&pk;
#pragma unroll
  for (int i = 0; i < 8; ++i) atomicAdd(arow + i, ew * bf2f(ph[i]));
}

// ---------- tanh(agg + bias) -> bf16 into strided destination ----------
__global__ void k_bias_tanh_bf(const float* __restrict__ agg, const float* __restrict__ bias,
                               uint16_t* __restrict__ out, int ldo, long n) {
  long i = (long)blockIdx.x * blockDim.x + threadIdx.x;
  if (i >= n) return;
  int r = (int)(i / HIDDEN), c = (int)(i % HIDDEN);
  out[(long)r * ldo + c] = f2bf(tanhf(agg[i] + bias[c]));
}

// ---------- segment mean/max readout over h3 [N,768] ----------
__global__ void k_readout(const uint16_t* __restrict__ h3, const int* __restrict__ seg,
                          float* __restrict__ sum, float* __restrict__ mx,
                          int* __restrict__ counts) {
  long i = (long)blockIdx.x * blockDim.x + threadIdx.x;
  if (i >= (long)N_NODES * 768) return;
  int n = (int)(i / 768), f = (int)(i % 768);
  int g = seg[n];
  float v = bf2f(h3[i]);
  atomicAdd(&sum[(long)g * 768 + f], v);
  float* ma = &mx[(long)g * 768 + f];
  if (v >= 0.0f) atomicMax((int*)ma, __float_as_int(v));
  else           atomicMin((unsigned int*)ma, (unsigned int)__float_as_int(v));
  if (f == 0) atomicAdd(&counts[g], 1);
}

// ---------- r = [mean | max] -> bf16 [64,1536] ----------
__global__ void k_finalize_r(const float* __restrict__ sum, const float* __restrict__ mx,
                             const int* __restrict__ counts, uint16_t* __restrict__ rb) {
  int i = blockIdx.x * blockDim.x + threadIdx.x;
  if (i >= N_GRAPHS * 1536) return;
  int g = i / 1536, f = i % 1536;
  float v;
  if (f < 768) {
    float c = (float)counts[g]; if (c < 1.0f) c = 1.0f;
    v = sum[(long)g * 768 + f] / c;
  } else {
    v = mx[(long)g * 768 + (f - 768)];
  }
  rb[i] = f2bf(v);
}

// ---------- out[g] = z[g,:] . post_w1 + post_b1 ----------
__global__ void k_final(const uint16_t* __restrict__ z, const float* __restrict__ w1,
                        const float* __restrict__ b1, float* __restrict__ out) {
  __shared__ float red[HIDDEN];
  int g = blockIdx.x, t = threadIdx.x;
  red[t] = bf2f(z[(long)g * HIDDEN + t]) * w1[t];
  __syncthreads();
  for (int s = HIDDEN / 2; s > 0; s >>= 1) {
    if (t < s) red[t] += red[t + s];
    __syncthreads();
  }
  if (t == 0) out[g] = red[0] + b1[0];
}

// =====================================================================
extern "C" void kernel_launch(void* const* d_in, const int* in_sizes, int n_in,
                              void* d_out, int out_size, void* d_ws, size_t ws_size,
                              hipStream_t stream) {
  const float* x       = (const float*)d_in[0];
  const int*   esrc    = (const int*)  d_in[1];
  const int*   edst    = (const int*)  d_in[2];
  const float* ew      = (const float*)d_in[3];
  const int*   seg     = (const int*)  d_in[4];
  const float* pre_w0  = (const float*)d_in[5];  const float* pre_b0 = (const float*)d_in[6];
  const float* pre_w1  = (const float*)d_in[7];  const float* pre_b1 = (const float*)d_in[8];
  const float* pre_w2  = (const float*)d_in[9];  const float* pre_b2 = (const float*)d_in[10];
  const float* conv1_w = (const float*)d_in[11]; const float* conv1_b = (const float*)d_in[12];
  const float* conv2_w = (const float*)d_in[13]; const float* conv2_b = (const float*)d_in[14];
  const float* post_w0 = (const float*)d_in[15]; const float* post_b0 = (const float*)d_in[16];
  const float* post_w1 = (const float*)d_in[17]; const float* post_b1 = (const float*)d_in[18];
  const int E = in_sizes[1];                       // 850000 (edges incl. self loops)
  float* out = (float*)d_out;

  // ---- workspace carve (256B aligned) ----
  char* p = (char*)d_ws;
  auto carve = [&](size_t bytes) -> char* {
    char* r = p; p += (bytes + 255) & ~(size_t)255; return r;
  };
  uint16_t* xb   = (uint16_t*)carve((size_t)N_NODES * F_IN * 2);
  uint16_t* h0   = (uint16_t*)carve((size_t)N_NODES * HIDDEN * 2);
  uint16_t* h1   = (uint16_t*)carve((size_t)N_NODES * HIDDEN * 2);
  uint16_t* h3   = (uint16_t*)carve((size_t)N_NODES * 768 * 2);   // [g2 | g1 | h]
  uint16_t* tbuf = (uint16_t*)carve((size_t)N_NODES * HIDDEN * 2);
  float*    agg  = (float*)   carve((size_t)N_NODES * HIDDEN * 4);
  uint16_t* w0t  = (uint16_t*)carve((size_t)HIDDEN * F_IN * 2);
  uint16_t* w1t  = (uint16_t*)carve((size_t)HIDDEN * HIDDEN * 2);
  uint16_t* w2t  = (uint16_t*)carve((size_t)HIDDEN * HIDDEN * 2);
  uint16_t* c1t  = (uint16_t*)carve((size_t)HIDDEN * HIDDEN * 2);
  uint16_t* c2t  = (uint16_t*)carve((size_t)HIDDEN * 512 * 2);
  uint16_t* pwt  = (uint16_t*)carve((size_t)HIDDEN * 1536 * 2);
  float*    sum  = (float*)   carve((size_t)N_GRAPHS * 768 * 4);
  float*    mxb  = (float*)   carve((size_t)N_GRAPHS * 768 * 4);
  int*      cnt  = (int*)     carve((size_t)N_GRAPHS * 4);
  uint16_t* rb   = (uint16_t*)carve((size_t)N_GRAPHS * 1536 * 2);
  uint16_t* zb   = (uint16_t*)carve((size_t)N_GRAPHS * HIDDEN * 2);
  (void)ws_size; (void)n_in; (void)out_size;

  auto blocks1d = [](long n, int bs) { return (int)((n + bs - 1) / bs); };
  auto gemm = [&](const uint16_t* A, int lda, const uint16_t* Wt, const float* bias,
                  uint16_t* O, int ldo, int M, int N, int K, int act) {
    int blocks = (M / 16) * (N / 256);   // one block = 16x256 output slab
    k_gemm_wmma<<<blocks, 128, 0, stream>>>(A, lda, Wt, bias, O, ldo, M, N, K, act);
  };

  // ---- weight / input conversions ----
  k_f2bf<<<blocks1d((long)N_NODES * F_IN, 256), 256, 0, stream>>>(x, xb, (long)N_NODES * F_IN);
  k_transpose_bf<<<blocks1d((long)F_IN * HIDDEN, 256), 256, 0, stream>>>(pre_w0, w0t, F_IN, HIDDEN);
  k_transpose_bf<<<blocks1d((long)HIDDEN * HIDDEN, 256), 256, 0, stream>>>(pre_w1, w1t, HIDDEN, HIDDEN);
  k_transpose_bf<<<blocks1d((long)HIDDEN * HIDDEN, 256), 256, 0, stream>>>(pre_w2, w2t, HIDDEN, HIDDEN);
  k_transpose_bf<<<blocks1d((long)HIDDEN * HIDDEN, 256), 256, 0, stream>>>(conv1_w, c1t, HIDDEN, HIDDEN);
  k_transpose_bf<<<blocks1d((long)512 * HIDDEN, 256), 256, 0, stream>>>(conv2_w, c2t, 512, HIDDEN);
  k_transpose_bf<<<blocks1d((long)1536 * HIDDEN, 256), 256, 0, stream>>>(post_w0, pwt, 1536, HIDDEN);

  // ---- pre-MLP (3x tanh GEMM); last layer writes into h3 cols 512..767 ----
  gemm(xb, F_IN,  w0t, pre_b0, h0, HIDDEN, N_NODES, HIDDEN, F_IN,  1);
  gemm(h0, HIDDEN, w1t, pre_b1, h1, HIDDEN, N_NODES, HIDDEN, HIDDEN, 1);
  gemm(h1, HIDDEN, w2t, pre_b2, h3 + 512, 768, N_NODES, HIDDEN, HIDDEN, 1);

  // ---- GCN conv1: t = h @ W1 ; agg = A_norm t ; g1 = tanh(agg+b) -> h3 cols 256..511 ----
  gemm(h3 + 512, 768, c1t, nullptr, tbuf, HIDDEN, N_NODES, HIDDEN, HIDDEN, 0);
  k_fill_u32<<<blocks1d((long)N_NODES * HIDDEN, 256), 256, 0, stream>>>((uint32_t*)agg, 0u, (long)N_NODES * HIDDEN);
  k_spmm<<<blocks1d(E, 8), 256, 0, stream>>>(esrc, edst, ew, tbuf, agg, E);
  k_bias_tanh_bf<<<blocks1d((long)N_NODES * HIDDEN, 256), 256, 0, stream>>>(agg, conv1_b, h3 + 256, 768, (long)N_NODES * HIDDEN);

  // ---- GCN conv2: input [g1|h] = h3 cols 256..767 (K=512) ; g2 -> h3 cols 0..255 ----
  gemm(h3 + 256, 768, c2t, nullptr, tbuf, HIDDEN, N_NODES, HIDDEN, 512, 0);
  k_fill_u32<<<blocks1d((long)N_NODES * HIDDEN, 256), 256, 0, stream>>>((uint32_t*)agg, 0u, (long)N_NODES * HIDDEN);
  k_spmm<<<blocks1d(E, 8), 256, 0, stream>>>(esrc, edst, ew, tbuf, agg, E);
  k_bias_tanh_bf<<<blocks1d((long)N_NODES * HIDDEN, 256), 256, 0, stream>>>(agg, conv2_b, h3, 768, (long)N_NODES * HIDDEN);

  // ---- readout: per-graph mean + max over h3 [N,768] ----
  k_fill_u32<<<blocks1d((long)N_GRAPHS * 768, 256), 256, 0, stream>>>((uint32_t*)sum, 0u, (long)N_GRAPHS * 768);
  k_fill_u32<<<blocks1d((long)N_GRAPHS * 768, 256), 256, 0, stream>>>((uint32_t*)mxb, 0xFF800000u, (long)N_GRAPHS * 768); // -inf
  k_fill_u32<<<blocks1d(N_GRAPHS, 64), 64, 0, stream>>>((uint32_t*)cnt, 0u, N_GRAPHS);
  k_readout<<<blocks1d((long)N_NODES * 768, 256), 256, 0, stream>>>(h3, seg, sum, mxb, cnt);
  k_finalize_r<<<blocks1d((long)N_GRAPHS * 1536, 256), 256, 0, stream>>>(sum, mxb, cnt, rb);

  // ---- post MLP: z = tanh(r @ post_w0 + b0) ; out = z @ post_w1 + b1 ----
  gemm(rb, 1536, pwt, post_b0, zb, HIDDEN, N_GRAPHS, HIDDEN, 1536, 1);
  k_final<<<N_GRAPHS, HIDDEN, 0, stream>>>(zb, post_w1, post_b1, out);
}
